// WLGNN_d_5549097746505
// MI455X (gfx1250) — matrix operand order repaired and checked by reading, not compiled, not run on hardware
//
#include <hip/hip_runtime.h>
#include <math.h>

// ---------------------------------------------------------------------------
// CDNA5 (gfx1250) WMMA types
// ---------------------------------------------------------------------------
typedef __attribute__((ext_vector_type(16))) __bf16 v16bf;
typedef __attribute__((ext_vector_type(8)))  float  v8f;

__device__ __forceinline__ v8f wmma_bf16(v16bf a, v16bf b, v8f c) {
    // D = A(16x32 bf16) * B(32x16 bf16) + C(16x16 f32)
    return __builtin_amdgcn_wmma_f32_16x16x32_bf16(
        /*neg_a=*/false, a, /*neg_b=*/false, b,
        /*c_mod=*/(short)0, c, /*reuse_a=*/false, /*reuse_b=*/false);
}

// Async copy 16B granules from global to LDS (ASYNCcnt path), per-lane issue.
__device__ __forceinline__ void async_g2lds_b128(unsigned lds_byte_off,
                                                 const void* gaddr) {
    unsigned long long ga = (unsigned long long)(uintptr_t)gaddr;
    asm volatile("global_load_async_to_lds_b128 %0, %1, off"
                 :: "v"(lds_byte_off), "v"(ga) : "memory");
}
__device__ __forceinline__ void wait_async0() {
    asm volatile("s_wait_asynccnt 0" ::: "memory");
}

// ---------------------------------------------------------------------------
// Utility kernels
// ---------------------------------------------------------------------------
__global__ void fill_zero_u32(unsigned* __restrict__ p, long long n) {
    long long i  = (long long)blockIdx.x * blockDim.x + threadIdx.x;
    long long st = (long long)gridDim.x * blockDim.x;
    for (; i < n; i += st) p[i] = 0u;
}

__global__ void deg_count(const int* __restrict__ dst, unsigned* __restrict__ cnt_d,
                          const int* __restrict__ dst_r, unsigned* __restrict__ cnt_r,
                          int E) {
    int e = blockIdx.x * blockDim.x + threadIdx.x;
    if (e >= E) return;
    atomicAdd(&cnt_d[dst[e]], 1u);
    atomicAdd(&cnt_r[dst_r[e]], 1u);
}

__global__ void deg_rsqrt(const unsigned* __restrict__ cnt, float* __restrict__ dis, int N) {
    int i = blockIdx.x * blockDim.x + threadIdx.x;
    if (i < N) dis[i] = rsqrtf((float)(cnt[i] + 1u));   // deg = incoming + self-loop
}

// ---------------------------------------------------------------------------
// GEMM: C[M x 32] = A[M x K] @ W[K x 32]     (K = 128 or 32, M multiple of 16)
// fp32 via bf16 hi/lo split (hi*hi + hi*lo + lo*hi -> ~fp32 accuracy).
// W is async-copied to LDS, then pre-split once into bf16 hi/lo fragments laid
// out in WMMA B order, so the k-loop is ds_load_b128 pairs + 3 WMMAs per tile.
// ---------------------------------------------------------------------------
__global__ __launch_bounds__(256)
void gemm_n32_wmma(const float* __restrict__ A, const float* __restrict__ W,
                   float* __restrict__ C, int Mtiles, int K) {
    __shared__ alignas(16) float  sStage[128 * 32];   // raw f32 W (async target)
    __shared__ alignas(32) __bf16 sWhi[128 * 32];     // fragment-ordered hi
    __shared__ alignas(32) __bf16 sWlo[128 * 32];     // fragment-ordered lo
    const int tid = threadIdx.x;
    const int tot = K * 32;

    // Stage W -> LDS through the async pipe (16B per transfer).
    {
        unsigned lds0 = (unsigned)(uintptr_t)(void*)sStage;  // as3 offset in low bits
        int nvec = tot >> 2;
        for (int i = tid; i < nvec; i += 256)
            async_g2lds_b128(lds0 + (unsigned)i * 16u, W + i * 4);
        wait_async0();                                        // this wave's transfers
    }
    __syncthreads();                                          // all waves' transfers

    // One-time split into WMMA B-fragment order:
// idx(ktb,nt,kh,m,e) = (((ktb*2+nt)*2+kh)*16+m)*16+e ; src k=ktb*32+(e&7)+kh*8+(e>>3)*16, col=nt*16+m
    for (int f = tid; f < tot; f += 256) {
        int e = f & 15, m = (f >> 4) & 15, kh = (f >> 8) & 1, nt = (f >> 9) & 1, ktb = f >> 10;
        float w = sStage[(ktb * 32 + (e & 7) + (kh << 3) + ((e >> 3) << 4)) * 32 + nt * 16 + m];
        __bf16 hi = (__bf16)w;
        sWhi[f] = hi;
        sWlo[f] = (__bf16)(w - (float)hi);
    }
    __syncthreads();

    const int wave = tid >> 5, lane = tid & 31;
    const int rowTile = blockIdx.x * 8 + wave;
    if (rowTile >= Mtiles) return;          // wave-uniform: EXEC all-1s at WMMA

    const int kh = lane >> 4;               // K-half held by this lane
    const int m  = lane & 15;               // A row / B,C col within tile
    const float* arow = A + (size_t)(rowTile * 16 + m) * K;

    v8f acc0 = {}; v8f acc1 = {};
    for (int kt = 0, ktb = 0; kt < K; kt += 32, ++ktb) {
        // A fragment: e<8 -> 8 contiguous floats at kt+kh*8; e>=8 -> at kt+16+kh*8
        const float4* ar0 = (const float4*)(arow + kt + (kh << 3));
        const float4* ar1 = (const float4*)(arow + kt + 16 + (kh << 3));
        float4 a0 = ar0[0], a1 = ar0[1], a2 = ar1[0], a3 = ar1[1];
        float af[16] = {a0.x, a0.y, a0.z, a0.w, a1.x, a1.y, a1.z, a1.w,
                        a2.x, a2.y, a2.z, a2.w, a3.x, a3.y, a3.z, a3.w};
        v16bf ahi, alo;
#pragma unroll
        for (int e = 0; e < 16; ++e) {
            __bf16 h = (__bf16)af[e];
            ahi[e] = h;
            alo[e] = (__bf16)(af[e] - (float)h);
        }
#pragma unroll
        for (int nt = 0; nt < 2; ++nt) {
            int base = (((ktb * 2 + nt) * 2 + kh) * 16 + m) * 16;
            v16bf bhi = *(const v16bf*)(sWhi + base);   // 32B lane-contiguous
            v16bf blo = *(const v16bf*)(sWlo + base);
            v8f acc = nt ? acc1 : acc0;
            acc = wmma_bf16(ahi, bhi, acc);   // hi*hi
            acc = wmma_bf16(ahi, blo, acc);   // hi*lo
            acc = wmma_bf16(alo, bhi, acc);   // lo*hi
            if (nt) acc1 = acc; else acc0 = acc;
        }
    }
    // C/D layout: VGPR r -> row r (lanes 0-15) / r+8 (lanes 16-31), col = lane&15
    const int r0 = rowTile * 16 + (kh << 3);
#pragma unroll
    for (int r = 0; r < 8; ++r) {
        C[(size_t)(r0 + r) * 32 + m]      = acc0[r];
        C[(size_t)(r0 + r) * 32 + m + 16] = acc1[r];
    }
}

// ---------------------------------------------------------------------------
// Edge aggregation: agg[dst] += dis[src]*dis[dst] * h[src]
// 8 threads per edge, float4 per thread; h (25.6MB) is L2-resident on MI455X.
// ---------------------------------------------------------------------------
__global__ __launch_bounds__(256)
void edge_agg(const int* __restrict__ src, const int* __restrict__ dst,
              const float* __restrict__ dis, const float* __restrict__ h,
              float* __restrict__ agg, int E) {
    int gid  = blockIdx.x * blockDim.x + threadIdx.x;
    int e    = gid >> 3;
    int part = gid & 7;
    if (e >= E) return;
    if (part == 0) {                         // stream the edge lists ahead
        __builtin_prefetch(src + e + 8192, 0, 0);
        __builtin_prefetch(dst + e + 8192, 0, 0);
    }
    int s = src[e], d = dst[e];
    float coef = dis[s] * dis[d];
    float4 hv = ((const float4*)(h + (size_t)s * 32))[part];
    float* ap = agg + (size_t)d * 32 + part * 4;
    atomicAdd(ap + 0, coef * hv.x);
    atomicAdd(ap + 1, coef * hv.y);
    atomicAdd(ap + 2, coef * hv.z);
    atomicAdd(ap + 3, coef * hv.w);
}

// ---------------------------------------------------------------------------
// Fused: (agg + deg^-1 * h + bias) -> LayerNorm -> ReLU for both branches,
// then sum. One thread per node (32 channels in registers).
// ---------------------------------------------------------------------------
__global__ __launch_bounds__(256)
void combine_ln(const float* __restrict__ aggA, const float* __restrict__ hA,
                const float* __restrict__ disA, const float* __restrict__ biasA,
                const float* __restrict__ aggB, const float* __restrict__ hB,
                const float* __restrict__ disB, const float* __restrict__ biasB,
                float* __restrict__ out, int N) {
    int i = blockIdx.x * blockDim.x + threadIdx.x;
    if (i >= N) return;
    float ia = disA[i] * disA[i];            // 1/deg (self-loop norm)
    float ib = disB[i] * disB[i];
    const float* pa = aggA + (size_t)i * 32;
    const float* ha = hA   + (size_t)i * 32;
    const float* pb = aggB + (size_t)i * 32;
    const float* hb = hB   + (size_t)i * 32;
    float va[32], vb[32];
    float sa = 0.f, sa2 = 0.f, sb = 0.f, sb2 = 0.f;
#pragma unroll
    for (int c = 0; c < 32; ++c) {
        float v = pa[c] + ia * ha[c] + biasA[c];
        va[c] = v; sa += v; sa2 += v * v;
        float w = pb[c] + ib * hb[c] + biasB[c];
        vb[c] = w; sb += w; sb2 += w * w;
    }
    float mua = sa * (1.f / 32.f), mub = sb * (1.f / 32.f);
    float ra = rsqrtf(fmaxf(sa2 * (1.f / 32.f) - mua * mua, 0.f) + 1e-5f);
    float rb = rsqrtf(fmaxf(sb2 * (1.f / 32.f) - mub * mub, 0.f) + 1e-5f);
    float* o = out + (size_t)i * 32;
#pragma unroll
    for (int c = 0; c < 32; ++c) {
        float ya = (va[c] - mua) * ra; ya = ya > 0.f ? ya : 0.f;
        float yb = (vb[c] - mub) * rb; yb = yb > 0.f ? yb : 0.f;
        o[c] = ya + yb;
    }
}

__global__ void fold_kernel(const float* __restrict__ hL, float* __restrict__ g,
                            int n, int off) {
    int i = blockIdx.x * blockDim.x + threadIdx.x;
    if (i < n) g[i] = hL[i] + hL[i + off];
}

// ---------------------------------------------------------------------------
// Head: out[Mh x 100] = sigmoid(G[Mh x 32] @ Wm[32 x 100] + bm)
// 7 column tiles (112 padded, masked at col>=100). One wave per 16x16 tile.
// ---------------------------------------------------------------------------
__global__ __launch_bounds__(256)
void head_gemm_wmma(const float* __restrict__ G, const float* __restrict__ Wm,
                    const float* __restrict__ bm, float* __restrict__ out, int Mtiles) {
    const int tid = threadIdx.x;
    const int wave = tid >> 5, lane = tid & 31;
    const int t = blockIdx.x * 8 + wave;
    const int mt = t / 7, nt = t % 7;
    if (mt >= Mtiles) return;                // wave-uniform
    const int kh = lane >> 4, m = lane & 15;

    const float* arow = G + (size_t)(mt * 16 + m) * 32;
    const float4* ar0 = (const float4*)(arow + (kh << 3));
    const float4* ar1 = (const float4*)(arow + 16 + (kh << 3));
    float4 a0 = ar0[0], a1 = ar0[1], a2 = ar1[0], a3 = ar1[1];
    float af[16] = {a0.x, a0.y, a0.z, a0.w, a1.x, a1.y, a1.z, a1.w,
                    a2.x, a2.y, a2.z, a2.w, a3.x, a3.y, a3.z, a3.w};
    v16bf ahi, alo;
#pragma unroll
    for (int e = 0; e < 16; ++e) {
        __bf16 h = (__bf16)af[e];
        ahi[e] = h; alo[e] = (__bf16)(af[e] - (float)h);
    }
    const int col = nt * 16 + m;
    v16bf bhi, blo;
#pragma unroll
    for (int e = 0; e < 16; ++e) {
        int k = (e & 7) + (kh << 3) + ((e >> 3) << 4);
        float w = (col < 100) ? Wm[k * 100 + col] : 0.f;
        __bf16 h = (__bf16)w;
        bhi[e] = h; blo[e] = (__bf16)(w - (float)h);
    }
    v8f acc = {};
    acc = wmma_bf16(ahi, bhi, acc);
    acc = wmma_bf16(ahi, blo, acc);
    acc = wmma_bf16(alo, bhi, acc);

    if (col < 100) {
        float bias = bm[col];
        const int r0 = mt * 16 + (kh << 3);
#pragma unroll
        for (int r = 0; r < 8; ++r) {
            float v = acc[r] + bias;
            out[(size_t)(r0 + r) * 100 + col] = 1.f / (1.f + expf(-v));
        }
    }
}

// ---------------------------------------------------------------------------
// Host launcher
// ---------------------------------------------------------------------------
extern "C" void kernel_launch(void* const* d_in, const int* in_sizes, int n_in,
                              void* d_out, int out_size, void* d_ws, size_t ws_size,
                              hipStream_t stream) {
    (void)n_in; (void)out_size; (void)ws_size;
    const float* x     = (const float*)d_in[0];
    const int*   src   = (const int*)d_in[1];
    const int*   dst   = (const int*)d_in[2];
    const int*   src_r = (const int*)d_in[3];
    const int*   dst_r = (const int*)d_in[4];
    const float* W1  = (const float*)d_in[5];
    const float* b1  = (const float*)d_in[6];
    const float* W2  = (const float*)d_in[7];
    const float* b2  = (const float*)d_in[8];
    const float* Wr1 = (const float*)d_in[9];
    const float* br1 = (const float*)d_in[10];
    const float* Wr2 = (const float*)d_in[11];
    const float* br2 = (const float*)d_in[12];
    const float* Wm  = (const float*)d_in[13];
    const float* bm  = (const float*)d_in[14];
    float* out = (float*)d_out;

    const int N  = in_sizes[0] / 128;   // 200000
    const int E  = in_sizes[1];         // 6400000
    const int NH = N / 2;

    // Carve workspace
    char* p = (char*)d_ws;
    auto take = [&](size_t bytes) -> char* {
        char* r = p; p += (bytes + 255) & ~(size_t)255; return r;
    };
    unsigned* cnt_d = (unsigned*)take((size_t)N * 4);
    unsigned* cnt_r = (unsigned*)take((size_t)N * 4);
    float* dis_d = (float*)take((size_t)N * 4);
    float* dis_r = (float*)take((size_t)N * 4);
    float* h_d   = (float*)take((size_t)N * 32 * 4);
    float* h_r   = (float*)take((size_t)N * 32 * 4);
    float* agg_d = (float*)take((size_t)N * 32 * 4);
    float* agg_r = (float*)take((size_t)N * 32 * 4);
    float* hL    = (float*)take((size_t)N * 32 * 4);
    float* g     = (float*)take((size_t)NH * 32 * 4);

    const int B = 256;
    const int Mtiles = N / 16;
    const int gemmBlocks = (Mtiles + 7) / 8;
    const int aggBlocks  = (int)(((long long)E * 8 + B - 1) / B);
    const int nodeBlocks = (N + B - 1) / B;

    // Degrees (shared by both layers)
    fill_zero_u32<<<512, B, 0, stream>>>(cnt_d, N);
    fill_zero_u32<<<512, B, 0, stream>>>(cnt_r, N);
    deg_count<<<(E + B - 1) / B, B, 0, stream>>>(dst, cnt_d, dst_r, cnt_r, E);
    deg_rsqrt<<<nodeBlocks, B, 0, stream>>>(cnt_d, dis_d, N);
    deg_rsqrt<<<nodeBlocks, B, 0, stream>>>(cnt_r, dis_r, N);

    // ---- Layer 1 ----
    gemm_n32_wmma<<<gemmBlocks, B, 0, stream>>>(x, W1,  h_d, Mtiles, 128);
    gemm_n32_wmma<<<gemmBlocks, B, 0, stream>>>(x, Wr1, h_r, Mtiles, 128);
    fill_zero_u32<<<2048, B, 0, stream>>>((unsigned*)agg_d, (long long)N * 32);
    fill_zero_u32<<<2048, B, 0, stream>>>((unsigned*)agg_r, (long long)N * 32);
    edge_agg<<<aggBlocks, B, 0, stream>>>(src,   dst,   dis_d, h_d, agg_d, E);
    edge_agg<<<aggBlocks, B, 0, stream>>>(src_r, dst_r, dis_r, h_r, agg_r, E);
    combine_ln<<<nodeBlocks, B, 0, stream>>>(agg_d, h_d, dis_d, b1,
                                             agg_r, h_r, dis_r, br1, hL, N);

    // ---- Layer 2 ----
    gemm_n32_wmma<<<gemmBlocks, B, 0, stream>>>(hL, W2,  h_d, Mtiles, 32);
    gemm_n32_wmma<<<gemmBlocks, B, 0, stream>>>(hL, Wr2, h_r, Mtiles, 32);
    fill_zero_u32<<<2048, B, 0, stream>>>((unsigned*)agg_d, (long long)N * 32);
    fill_zero_u32<<<2048, B, 0, stream>>>((unsigned*)agg_r, (long long)N * 32);
    edge_agg<<<aggBlocks, B, 0, stream>>>(src,   dst,   dis_d, h_d, agg_d, E);
    edge_agg<<<aggBlocks, B, 0, stream>>>(src_r, dst_r, dis_r, h_r, agg_r, E);
    combine_ln<<<nodeBlocks, B, 0, stream>>>(agg_d, h_d, dis_d, b2,
                                             agg_r, h_r, dis_r, br2, hL, N);

    // ---- Fold + head ----
    fold_kernel<<<(NH * 32 + B - 1) / B, B, 0, stream>>>(hL, g, NH * 32, NH * 32);
    const int Mh = NH / 16;
    head_gemm_wmma<<<(Mh * 7 + 7) / 8, B, 0, stream>>>(g, Wm, bm, out, Mh);
}
